// MultiHeadSelfAttention_37606733644520
// MI455X (gfx1250) — matrix operand order repaired and checked by reading, not compiled, not run on hardware
//
#include <hip/hip_runtime.h>
#include <hip/hip_bf16.h>
#include <stdint.h>

#define B_ 4
#define C_ 2048
#define E_ 1024
#define H_ 16
#define D_ 64

typedef __attribute__((ext_vector_type(16))) __bf16 bf16x16;
typedef __attribute__((ext_vector_type(8)))  float  v8f;
typedef __attribute__((ext_vector_type(4)))  int    v4i;

struct __attribute__((aligned(16))) U4 { uint32_t x, y, z, w; };
struct __attribute__((aligned(8)))  U2 { uint32_t x, y; };

union Frag {
    bf16x16        v;
    U4             u[2];
    unsigned short s[16];
};

#if defined(__HIP_DEVICE_COMPILE__) && defined(__has_builtin)
#if __has_builtin(__builtin_amdgcn_global_load_async_to_lds_b128) && \
    __has_builtin(__builtin_amdgcn_s_wait_asynccnt)
#define USE_ASYNC_LDS 1
#endif
#endif
#ifndef USE_ASYNC_LDS
#define USE_ASYNC_LDS 0
#endif

#if USE_ASYNC_LDS
// Param types from the compiler diagnostic: pointer to 16-byte int vector,
// global (AS1) source / LDS (AS3) destination.
typedef __attribute__((address_space(1))) v4i* gv4p;
typedef __attribute__((address_space(3))) v4i* lv4p;
__device__ __forceinline__ void async_cp16(const unsigned short* g, unsigned short* l) {
    __builtin_amdgcn_global_load_async_to_lds_b128((gv4p)g, (lv4p)l, 0, 0);
}
#endif

__device__ __forceinline__ unsigned short f2bf(float f) {
    union { float f; uint32_t u; } x;
    x.f = f;
    uint32_t r = x.u + 0x7FFFu + ((x.u >> 16) & 1u);  // round-to-nearest-even
    return (unsigned short)(r >> 16);
}
__device__ __forceinline__ uint32_t pack2bf(float a, float b) {
    return (uint32_t)f2bf(a) | ((uint32_t)f2bf(b) << 16);
}

// ---------------------------------------------------------------------------
// NT GEMM: C[M,N] = A[M,K] * B[N,K]^T   (A f32 or bf16-raw, B f32, out f32/bf16)
// 256 threads (8 waves), tile 128x128, BK=32, double-buffered LDS,
// software-pipelined fetch(global->regs) / compute / commit(regs->LDS).
// ---------------------------------------------------------------------------
template <bool A_IS_BF16, bool OUT_BF16>
__global__ __launch_bounds__(256) void gemm_nt(const void* __restrict__ Ap,
                                               const float* __restrict__ Bp,
                                               void* __restrict__ Cp,
                                               int M, int N, int K) {
    constexpr int BM = 128, BN = 128, BK = 32, LDT = BK + 8;  // 40-short rows
    __shared__ unsigned short lA[2][BM * LDT];
    __shared__ unsigned short lB[2][BN * LDT];

    const int t    = threadIdx.x;
    const int wid  = t >> 5;
    const int lane = t & 31;
    const int hl   = lane >> 4;
    const int l16  = lane & 15;

    const int m0 = blockIdx.y * BM;
    const int n0 = blockIdx.x * BN;
    const int wm = (wid & 3) * 32;
    const int wn = (wid >> 2) * 64;

    const int lr = t >> 3;        // 0..31 (8 threads per 32-wide row)
    const int lc = (t & 7) * 4;

    float4 fA[4], fB[4];
    U2     rA[4];

    auto fetch = [&](int kk) {
#pragma unroll
        for (int p = 0; p < 4; ++p) {
            int r = lr + p * 32;
            if constexpr (A_IS_BF16)
                rA[p] = *(const U2*)((const unsigned short*)Ap +
                                     (size_t)(m0 + r) * K + kk + lc);
            else
                fA[p] = *(const float4*)((const float*)Ap +
                                         (size_t)(m0 + r) * K + kk + lc);
            fB[p] = *(const float4*)(Bp + (size_t)(n0 + r) * K + kk + lc);
        }
        if (kk + BK < K) {  // prefetch tile after next into L2 (global_prefetch_b8)
            __builtin_prefetch((const char*)(Bp + (size_t)(n0 + lr) * K + kk + BK + lc), 0, 0);
            if constexpr (!A_IS_BF16)
                __builtin_prefetch((const char*)((const float*)Ap +
                                   (size_t)(m0 + lr) * K + kk + BK + lc), 0, 0);
        }
    };
    auto commit = [&](int buf) {
#pragma unroll
        for (int p = 0; p < 4; ++p) {
            int r = lr + p * 32;
            if constexpr (A_IS_BF16) {
                *(U2*)&lA[buf][r * LDT + lc] = rA[p];
            } else {
                U2 d;
                d.x = pack2bf(fA[p].x, fA[p].y);
                d.y = pack2bf(fA[p].z, fA[p].w);
                *(U2*)&lA[buf][r * LDT + lc] = d;
            }
            U2 e;
            e.x = pack2bf(fB[p].x, fB[p].y);
            e.y = pack2bf(fB[p].z, fB[p].w);
            *(U2*)&lB[buf][r * LDT + lc] = e;
        }
    };

    v8f zero = {};
    v8f acc[2][4];
#pragma unroll
    for (int i = 0; i < 2; ++i)
#pragma unroll
        for (int j = 0; j < 4; ++j) acc[i][j] = zero;

    fetch(0);
    commit(0);
    __syncthreads();

    int buf = 0;
    for (int kk = 0; kk < K; kk += BK) {
        const bool more = (kk + BK) < K;
        if (more) fetch(kk + BK);  // global loads issue before compute

        Frag af[2], bfr[4];
#pragma unroll
        for (int mt = 0; mt < 2; ++mt) {
            // A layout: lane row = l16, K elems {hl*8..+7} and {hl*8+16..+7}
            const unsigned short* p = &lA[buf][(wm + mt * 16 + l16) * LDT + hl * 8];
            af[mt].u[0] = *(const U4*)p;
            af[mt].u[1] = *(const U4*)(p + 16);
        }
#pragma unroll
        for (int nt = 0; nt < 4; ++nt) {
            // B layout: lane col = l16, K elems {hl*8..+7} and {hl*8+16..+7}
            const unsigned short* p = &lB[buf][(wn + nt * 16 + l16) * LDT + hl * 8];
            bfr[nt].u[0] = *(const U4*)p;
            bfr[nt].u[1] = *(const U4*)(p + 16);
        }
#pragma unroll
        for (int mt = 0; mt < 2; ++mt)
#pragma unroll
            for (int nt = 0; nt < 4; ++nt)
                acc[mt][nt] = __builtin_amdgcn_wmma_f32_16x16x32_bf16(
                    false, af[mt].v, false, bfr[nt].v, (short)0, acc[mt][nt],
                    false, false);

        if (more) commit(buf ^ 1);  // loads have had the compute phase to land
        __syncthreads();
        buf ^= 1;
    }

#pragma unroll
    for (int mt = 0; mt < 2; ++mt)
#pragma unroll
        for (int nt = 0; nt < 4; ++nt)
#pragma unroll
            for (int i = 0; i < 8; ++i) {
                int r = m0 + wm + mt * 16 + i + hl * 8;  // C/D layout row
                int c = n0 + wn + nt * 16 + l16;         // C/D layout col
                if constexpr (OUT_BF16)
                    ((unsigned short*)Cp)[(size_t)r * N + c] = f2bf(acc[mt][nt][i]);
                else
                    ((float*)Cp)[(size_t)r * N + c] = acc[mt][nt][i];
            }
}

// ---------------------------------------------------------------------------
// Causal flash attention. qkv: bf16-raw (B*C, 3072) rows [Q|K|V] per token.
// grid = (C/128, B*H), block = 256 (8 waves), wave w owns q-rows bx*128+w*16..+15.
// Double-buffered K (async copy to LDS when available) and transposed V tiles.
// ---------------------------------------------------------------------------
__global__ __launch_bounds__(256) void flash_attn(
    const unsigned short* __restrict__ qkv, unsigned short* __restrict__ O) {
    constexpr int BQ = 128, BKV = 64, LDK = D_ + 8, LDV = BKV + 8;  // 72-short rows
    __shared__ unsigned short Kt[2][BKV * LDK];   // [key][d]
    __shared__ unsigned short VtT[2][D_ * LDV];   // transposed: [d][key]
    __shared__ unsigned short Pt[8][16 * LDK];

    const int t = threadIdx.x, wid = t >> 5, lane = t & 31;
    const int hl = lane >> 4, l16 = lane & 15;
    const int bh = blockIdx.y;
    const int b = bh >> 4, h = bh & (H_ - 1);
    const int q0 = blockIdx.x * BQ;

    const size_t rs = 3 * H_ * D_;  // 3072 shorts per token row
    const unsigned short* base = qkv + (size_t)b * C_ * rs;
    const int qcol = h * D_, kcol = E_ + h * D_, vcol = 2 * E_ + h * D_;

    // Q fragments: two K=32 steps covering d=0..63
    Frag qf[2];
    {
        int qr = q0 + wid * 16 + l16;
        const unsigned short* qp = base + (size_t)qr * rs + qcol;
#pragma unroll
        for (int st = 0; st < 2; ++st) {
            int kb = st * 32 + hl * 8;
            qf[st].u[0] = *(const U4*)(qp + kb);
            qf[st].u[1] = *(const U4*)(qp + kb + 16);
        }
    }

    v8f zero = {};
    v8f o[4];
#pragma unroll
    for (int dt = 0; dt < 4; ++dt) o[dt] = zero;
    float mi[8], li[8];
#pragma unroll
    for (int i = 0; i < 8; ++i) { mi[i] = -3.0e38f; li[i] = 0.f; }

    const float scale = 0.125f;                 // 1/sqrt(64)
    const int lr = t >> 2, lc = (t & 3) * 16;   // staging: 4 thr/row, 64 rows

    Frag vreg;         // fetched V segment (16 shorts), lives across compute
#if !USE_ASYNC_LDS
    U4 kreg[2];
#endif

    auto stage_fetch = [&](int kb0, int buf) {
        const unsigned short* kp = base + (size_t)(kb0 + lr) * rs + kcol + lc;
        const unsigned short* vp = base + (size_t)(kb0 + lr) * rs + vcol + lc;
#if USE_ASYNC_LDS
        unsigned short* ld = &Kt[buf][lr * LDK + lc];
        async_cp16(kp, ld);          // K straight to LDS, ASYNCcnt-tracked
        async_cp16(kp + 8, ld + 8);
#else
        (void)buf;
        kreg[0] = *(const U4*)kp;
        kreg[1] = *(const U4*)(kp + 8);
#endif
        vreg.u[0] = *(const U4*)vp;  // V must round-trip (transposed store)
        vreg.u[1] = *(const U4*)(vp + 8);
    };
    auto stage_commit = [&](int buf) {
#if !USE_ASYNC_LDS
        *(U4*)&Kt[buf][lr * LDK + lc]     = kreg[0];
        *(U4*)&Kt[buf][lr * LDK + lc + 8] = kreg[1];
#endif
#pragma unroll
        for (int j = 0; j < 16; ++j)  // transpose V: VtT[d][key]
            VtT[buf][(lc + j) * LDV + lr] = vreg.s[j];
    };

    const int kv_end = q0 + BQ;
    stage_fetch(0, 0);
    stage_commit(0);
#if USE_ASYNC_LDS
    __builtin_amdgcn_s_wait_asynccnt(0);
#endif
    __syncthreads();

    int buf = 0;
    for (int kb0 = 0; kb0 < kv_end; kb0 += BKV) {
        const bool more = (kb0 + BKV) < kv_end;
        if (more) stage_fetch(kb0 + BKV, buf ^ 1);  // overlap copy with compute

        // S = Q K^T  (4 key-tiles x 2 K-steps)
        v8f s[4];
#pragma unroll
        for (int nt = 0; nt < 4; ++nt) s[nt] = zero;
#pragma unroll
        for (int nt = 0; nt < 4; ++nt)
#pragma unroll
            for (int st = 0; st < 2; ++st) {
                Frag kf;
                const unsigned short* p =
                    &Kt[buf][(nt * 16 + l16) * LDK + st * 32 + hl * 8];
                kf.u[0] = *(const U4*)p;
                kf.u[1] = *(const U4*)(p + 16);
                s[nt] = __builtin_amdgcn_wmma_f32_16x16x32_bf16(
                    false, qf[st].v, false, kf.v, (short)0, s[nt], false, false);
            }

        // scale + causal mask + online softmax stats
        float mnew[8];
#pragma unroll
        for (int i = 0; i < 8; ++i) {
            int qr = q0 + wid * 16 + i + hl * 8;
            float mx = -3.0e38f;
#pragma unroll
            for (int nt = 0; nt < 4; ++nt) {
                int key = kb0 + nt * 16 + l16;
                float v = s[nt][i] * scale;
                if (key > qr) v = -3.0e38f;
                s[nt][i] = v;
                mx = fmaxf(mx, v);
            }
            mx = fmaxf(mx, __shfl_xor(mx, 1));
            mx = fmaxf(mx, __shfl_xor(mx, 2));
            mx = fmaxf(mx, __shfl_xor(mx, 4));
            mx = fmaxf(mx, __shfl_xor(mx, 8));
            mnew[i] = fmaxf(mi[i], mx);
        }
#pragma unroll
        for (int i = 0; i < 8; ++i) {
            float corr = __expf(mi[i] - mnew[i]);
            mi[i] = mnew[i];
            float rsum = 0.f;
#pragma unroll
            for (int nt = 0; nt < 4; ++nt) {
                float p = __expf(s[nt][i] - mnew[i]);
                s[nt][i] = p;
                rsum += p;
            }
            rsum += __shfl_xor(rsum, 1);
            rsum += __shfl_xor(rsum, 2);
            rsum += __shfl_xor(rsum, 4);
            rsum += __shfl_xor(rsum, 8);
            li[i] = li[i] * corr + rsum;
#pragma unroll
            for (int dt = 0; dt < 4; ++dt) o[dt][i] *= corr;
        }

        // P: C/D layout -> A layout via per-wave LDS scratch
#pragma unroll
        for (int i = 0; i < 8; ++i)
#pragma unroll
            for (int nt = 0; nt < 4; ++nt)
                Pt[wid][(i + hl * 8) * LDK + nt * 16 + l16] = f2bf(s[nt][i]);
        __builtin_amdgcn_wave_barrier();  // same-wave LDS ops are in-order
        Frag pa[2];
        {
            const unsigned short* pp = &Pt[wid][l16 * LDK];
#pragma unroll
            for (int st = 0; st < 2; ++st) {
                int kb = st * 32 + hl * 8;
                pa[st].u[0] = *(const U4*)(pp + kb);
                pa[st].u[1] = *(const U4*)(pp + kb + 16);
            }
        }

        // O += P V  (4 d-tiles x 2 K-steps over keys), V read from transposed tile
#pragma unroll
        for (int dt = 0; dt < 4; ++dt)
#pragma unroll
            for (int st = 0; st < 2; ++st) {
                Frag vf;
                const unsigned short* p =
                    &VtT[buf][(dt * 16 + l16) * LDV + st * 32 + hl * 8];
                vf.u[0] = *(const U4*)p;
                vf.u[1] = *(const U4*)(p + 16);
                o[dt] = __builtin_amdgcn_wmma_f32_16x16x32_bf16(
                    false, pa[st].v, false, vf.v, (short)0, o[dt], false, false);
            }

        if (more) stage_commit(buf ^ 1);
#if USE_ASYNC_LDS
        __builtin_amdgcn_s_wait_asynccnt(0);
#endif
        __syncthreads();
        buf ^= 1;
    }

    // epilogue: normalize and store bf16 O in (b, c, h*64+d) layout
#pragma unroll
    for (int i = 0; i < 8; ++i) {
        float inv = 1.0f / li[i];
        int qr = q0 + wid * 16 + i + hl * 8;
        unsigned short* op = O + ((size_t)b * C_ + qr) * (H_ * D_) + h * D_;
#pragma unroll
        for (int dt = 0; dt < 4; ++dt)
            op[dt * 16 + l16] = f2bf(o[dt][i] * inv);
    }
}

// ---------------------------------------------------------------------------
extern "C" void kernel_launch(void* const* d_in, const int* in_sizes, int n_in,
                              void* d_out, int out_size, void* d_ws,
                              size_t ws_size, hipStream_t stream) {
    (void)in_sizes; (void)n_in; (void)out_size; (void)ws_size;
    const float* x     = (const float*)d_in[0];
    const float* w_qkv = (const float*)d_in[1];
    const float* w_out = (const float*)d_in[2];

    unsigned short* qkv_ws = (unsigned short*)d_ws;                  // 8192x3072 bf16
    unsigned short* o_ws   = qkv_ws + (size_t)B_ * C_ * 3 * H_ * D_; // 8192x1024 bf16

    // 1) qkv = x @ w_qkv^T   (M=8192, N=3072, K=1024), f32 in -> bf16 out
    dim3 g1((3 * H_ * D_) / 128, (B_ * C_) / 128);
    gemm_nt<false, true><<<g1, 256, 0, stream>>>(x, w_qkv, qkv_ws,
                                                 B_ * C_, 3 * H_ * D_, E_);

    // 2) causal flash attention per (b,h)
    dim3 g2(C_ / 128, B_ * H_);
    flash_attn<<<g2, 256, 0, stream>>>(qkv_ws, o_ws);

    // 3) out = O @ w_out^T   (M=8192, N=1024, K=1024), bf16 A -> f32 out
    dim3 g3(E_ / 128, (B_ * C_) / 128);
    gemm_nt<true, false><<<g3, 256, 0, stream>>>(o_ws, w_out, d_out,
                                                 B_ * C_, E_, H_ * D_);
}
